// PointNet2SemSegTrans_14018773254177
// MI455X (gfx1250) — matrix operand order repaired and checked by reading, compiled, not run on hardware
//
#include <hip/hip_runtime.h>
#include <hip/hip_bf16.h>
#include <cmath>

typedef __attribute__((ext_vector_type(16))) _Float16 v16h;
typedef __attribute__((ext_vector_type(8)))  float    v8f;

#define WMMA_F16(A, B, Cacc) \
    __builtin_amdgcn_wmma_f32_16x16x32_f16(false, (A), false, (B), (short)0, (Cacc), false, false)

// =====================================================================
// Fast-path WMMA GEMM (O%32==0, Ncol%32==0):
//   Y[b][o][n] = sum_c W[o][c] * X[b][c][n] (+bias) (ReLU?)
// One wave32 computes a 32x32 output tile: 2 A fragments x 2 B fragments
// -> 4 accumulators, 4 v_wmma_f32_16x16x32_f16 per 32-wide K step.
// No edge guards anywhere: steady-state is pure load-clause/convert/WMMA.
// Only the ragged-K tail (C%32!=0) uses clamped loads + selects.
// =====================================================================
__global__ void k_gemm_wmma_f(const float* __restrict__ W, const float* __restrict__ X,
                              const float* __restrict__ bias, float* __restrict__ Y,
                              int Bb, int O, int C, int Ncol, int relu)
{
    const int tilesO = O >> 5;
    const int tilesN = Ncol >> 5;
    const int total  = Bb * tilesO * tilesN;
    const int wave   = (blockIdx.x * blockDim.x + threadIdx.x) >> 5;
    if (wave >= total) return;
    const int lane = threadIdx.x & 31;
    const int to   = wave % tilesO;
    const int rest = wave / tilesO;
    const int tn   = rest % tilesN;
    const int bb   = rest / tilesN;

    const int l15  = lane & 15;
    const int kgrp = (lane < 16) ? 0 : 8;            // A fragment K group per half-wave
    const float* __restrict__ Wr0 = W + (size_t)(to * 32 + l15) * C;
    const float* __restrict__ Wr1 = W + (size_t)(to * 32 + 16 + l15) * C;
    const int nb = tn * 32;
    const float* __restrict__ Xb = X + (size_t)bb * C * Ncol;

    v8f acc00 = {}, acc01 = {}, acc10 = {}, acc11 = {};
    int k0 = 0;
    for (; k0 + 32 <= C; k0 += 32) {
        v16h a0, a1, b0, b1;
        const float* wp0 = Wr0 + k0 + kgrp;          // two contiguous 8-float runs per row
        const float* wp1 = Wr1 + k0 + kgrp;
#pragma unroll
        for (int j = 0; j < 8; ++j) {
            a0[j]     = (_Float16)wp0[j];
            a0[8 + j] = (_Float16)wp0[16 + j];
            a1[j]     = (_Float16)wp1[j];
            a1[8 + j] = (_Float16)wp1[16 + j];
        }
        const float* xr = Xb + (size_t)k0 * Ncol + nb + l15;   // lane -> N column
#pragma unroll
        for (int j = 0; j < 16; ++j) {
            const int kk = (j < 8) ? (kgrp + j) : (kgrp + 8 + j);
            b0[j] = (_Float16)xr[(size_t)kk * Ncol];
            b1[j] = (_Float16)xr[(size_t)kk * Ncol + 16];
        }
        acc00 = WMMA_F16(a0, b0, acc00);
        acc01 = WMMA_F16(a0, b1, acc01);
        acc10 = WMMA_F16(a1, b0, acc10);
        acc11 = WMMA_F16(a1, b1, acc11);
    }
    if (k0 < C) {                                    // ragged-K tail: clamps + selects
        const int krem = C - k0;
        v16h a0, a1, b0, b1;
#pragma unroll
        for (int j = 0; j < 8; ++j) {
            int ko = kgrp + j;
            int kc = k0 + ((ko < krem) ? ko : (krem - 1));
            a0[j] = (_Float16)((ko < krem) ? Wr0[kc] : 0.0f);
            a1[j] = (_Float16)((ko < krem) ? Wr1[kc] : 0.0f);
            int ko2 = kgrp + 16 + j;
            int kc2 = k0 + ((ko2 < krem) ? ko2 : (krem - 1));
            a0[8 + j] = (_Float16)((ko2 < krem) ? Wr0[kc2] : 0.0f);
            a1[8 + j] = (_Float16)((ko2 < krem) ? Wr1[kc2] : 0.0f);
        }
        const float* xr = Xb + (size_t)k0 * Ncol + nb + l15;
#pragma unroll
        for (int j = 0; j < 16; ++j) {
            const int ko = (j < 8) ? (kgrp + j) : (kgrp + 8 + j);
            const bool kin = ko < krem;
            const float* xp = xr + (size_t)(kin ? ko : 0) * Ncol;
            float x0 = xp[0], x1 = xp[16];
            b0[j] = (_Float16)(kin ? x0 : 0.0f);
            b1[j] = (_Float16)(kin ? x1 : 0.0f);
        }
        acc00 = WMMA_F16(a0, b0, acc00);
        acc01 = WMMA_F16(a0, b1, acc01);
        acc10 = WMMA_F16(a1, b0, acc10);
        acc11 = WMMA_F16(a1, b1, acc11);
    }

    // store (C/D layout: VGPR r -> M = r + 8*(lane>=16), N = lane&15); no masks
    const int mofs = (lane < 16) ? 0 : 8;
    float* __restrict__ Yb = Y + (size_t)bb * O * Ncol;
    const int m0 = to * 32 + mofs;
    const int n0 = nb + l15, n1 = nb + 16 + l15;
#pragma unroll
    for (int r = 0; r < 8; ++r) {
        const int ma = m0 + r, mb2 = m0 + 16 + r;
        float v00 = acc00[r], v01 = acc01[r], v10 = acc10[r], v11 = acc11[r];
        if (bias) {
            float ba = bias[ma], bb2 = bias[mb2];
            v00 += ba; v01 += ba; v10 += bb2; v11 += bb2;
        }
        if (relu) {
            v00 = fmaxf(v00, 0.f); v01 = fmaxf(v01, 0.f);
            v10 = fmaxf(v10, 0.f); v11 = fmaxf(v11, 0.f);
        }
        Yb[(size_t)ma  * Ncol + n0] = v00;
        Yb[(size_t)ma  * Ncol + n1] = v01;
        Yb[(size_t)mb2 * Ncol + n0] = v10;
        Yb[(size_t)mb2 * Ncol + n1] = v11;
    }
}

// =====================================================================
// Generic WMMA GEMM (ragged O / Ncol): 16x32 tile, clamped edges.
// Used only for the 13-class head GEMM in this network.
// =====================================================================
__global__ void k_gemm_wmma(const float* __restrict__ W, const float* __restrict__ X,
                            const float* __restrict__ bias, float* __restrict__ Y,
                            int Bb, int O, int C, int Ncol, int relu)
{
    const int tilesO = (O + 15) >> 4;
    const int tilesN = (Ncol + 31) >> 5;
    const int total  = Bb * tilesO * tilesN;
    const int wave   = (blockIdx.x * blockDim.x + threadIdx.x) >> 5;
    if (wave >= total) return;
    const int lane = threadIdx.x & 31;
    const int to   = wave % tilesO;
    const int rest = wave / tilesO;
    const int tn   = rest % tilesN;
    const int b    = rest / tilesN;

    const int l15  = lane & 15;
    const int kgrp = (lane < 16) ? 0 : 8;
    const int mrow = to * 16 + l15;
    const int arow = (mrow < O) ? mrow : (O - 1);
    const float* __restrict__ Wr = W + (size_t)arow * C;

    const int nb = tn * 32;
    const bool hasN2 = (nb + 16) < Ncol;
    const float* __restrict__ Xb = X + (size_t)b * C * Ncol;

    v8f acc0 = {}, acc1 = {};
    int k0 = 0;
    for (; k0 + 32 <= C; k0 += 32) {
        v16h a, b0, b1;
        const float* wp = Wr + k0 + kgrp;
#pragma unroll
        for (int j = 0; j < 8; ++j) a[j]     = (_Float16)wp[j];
#pragma unroll
        for (int j = 0; j < 8; ++j) a[8 + j] = (_Float16)wp[16 + j];
        const float* xr = Xb + (size_t)k0 * Ncol + nb + l15;
#pragma unroll
        for (int j = 0; j < 16; ++j) b0[j] = (_Float16)xr[(size_t)((j < 8) ? (kgrp + j) : (kgrp + 8 + j)) * Ncol];
        if (hasN2) {
#pragma unroll
            for (int j = 0; j < 16; ++j) b1[j] = (_Float16)xr[(size_t)((j < 8) ? (kgrp + j) : (kgrp + 8 + j)) * Ncol + 16];
        } else {
#pragma unroll
            for (int j = 0; j < 16; ++j) b1[j] = (_Float16)0.0f;
        }
        acc0 = WMMA_F16(a, b0, acc0);
        acc1 = WMMA_F16(a, b1, acc1);
    }
    if (k0 < C) {
        const int krem = C - k0;
        v16h a, b0, b1;
#pragma unroll
        for (int j = 0; j < 8; ++j) {
            int ko = kgrp + j;
            float w = Wr[k0 + ((ko < krem) ? ko : (krem - 1))];
            a[j] = (_Float16)((ko < krem) ? w : 0.0f);
        }
#pragma unroll
        for (int j = 0; j < 8; ++j) {
            int ko = kgrp + 16 + j;
            float w = Wr[k0 + ((ko < krem) ? ko : (krem - 1))];
            a[8 + j] = (_Float16)((ko < krem) ? w : 0.0f);
        }
        const float* xr = Xb + (size_t)k0 * Ncol + nb + l15;
#pragma unroll
        for (int j = 0; j < 16; ++j) {
            const int ko = (j < 8) ? (kgrp + j) : (kgrp + 8 + j);
            const bool kin = ko < krem;
            float x = xr[(size_t)(kin ? ko : 0) * Ncol];
            b0[j] = (_Float16)(kin ? x : 0.0f);
        }
        if (hasN2) {
#pragma unroll
            for (int j = 0; j < 16; ++j) {
                const int ko = (j < 8) ? (kgrp + j) : (kgrp + 8 + j);
                const bool kin = ko < krem;
                float x = xr[(size_t)(kin ? ko : 0) * Ncol + 16];
                b1[j] = (_Float16)(kin ? x : 0.0f);
            }
        } else {
#pragma unroll
            for (int j = 0; j < 16; ++j) b1[j] = (_Float16)0.0f;
        }
        acc0 = WMMA_F16(a, b0, acc0);
        acc1 = WMMA_F16(a, b1, acc1);
    }

    const int mofs = (lane < 16) ? 0 : 8;
    float* __restrict__ Yb = Y + (size_t)b * O * Ncol;
    const int n0 = nb + l15;
#pragma unroll
    for (int r = 0; r < 8; ++r) {
        const int mm = to * 16 + mofs + r;
        if (mm < O) {
            float v = acc0[r];
            if (bias) v += bias[mm];
            if (relu) v = fmaxf(v, 0.0f);
            Yb[(size_t)mm * Ncol + n0] = v;
        }
    }
    if (hasN2) {
        const int n1 = nb + 16 + l15;
#pragma unroll
        for (int r = 0; r < 8; ++r) {
            const int mm = to * 16 + mofs + r;
            if (mm < O) {
                float v = acc1[r];
                if (bias) v += bias[mm];
                if (relu) v = fmaxf(v, 0.0f);
                Yb[(size_t)mm * Ncol + n1] = v;
            }
        }
    }
}

// ===================== BatchNorm (training-mode stats) =====================
__global__ void k_bn_stats(const float* __restrict__ X, float* __restrict__ mean,
                           float* __restrict__ var, int Bb, int O, int Ncol)
{
    const int o = blockIdx.x, tid = threadIdx.x;
    float s = 0.f, ss = 0.f;
    for (int b = 0; b < Bb; ++b) {
        const float* p = X + ((size_t)b * O + o) * Ncol;
        for (int n = tid; n < Ncol; n += blockDim.x) { float v = p[n]; s += v; ss += v * v; }
    }
    __shared__ float sh1[256], sh2[256];
    sh1[tid] = s; sh2[tid] = ss; __syncthreads();
    for (int st = 128; st > 0; st >>= 1) {
        if (tid < st) { sh1[tid] += sh1[tid + st]; sh2[tid] += sh2[tid + st]; }
        __syncthreads();
    }
    if (tid == 0) {
        float cnt = (float)Bb * (float)Ncol;
        float m = sh1[0] / cnt;
        mean[o] = m;
        var[o]  = sh2[0] / cnt - m * m;
    }
}

__global__ void k_bn_apply(float* __restrict__ X, const float* __restrict__ mean,
                           const float* __restrict__ var, const float* __restrict__ g,
                           const float* __restrict__ bta, int Bb, int O, int Ncol, int relu)
{
    size_t i = (size_t)blockIdx.x * blockDim.x + threadIdx.x;
    size_t total = (size_t)Bb * O * Ncol;
    if (i >= total) return;
    int o = (int)((i / (size_t)Ncol) % (size_t)O);
    float v = X[i];
    v = (v - mean[o]) * rsqrtf(var[o] + 1e-5f) * g[o] + bta[o];
    if (relu) v = fmaxf(v, 0.f);
    X[i] = v;
}

// ===================== elementwise =====================
__global__ void k_add(float* __restrict__ y, const float* __restrict__ x, size_t n)
{
    size_t i = (size_t)blockIdx.x * blockDim.x + threadIdx.x;
    if (i < n) y[i] += x[i];
}

// split points [B][9][N] -> xyz [B][3][N], feat [B][6][N]
__global__ void k_split(const float* __restrict__ pts, float* __restrict__ xyz,
                        float* __restrict__ f, int Bb, int N)
{
    int t = blockIdx.x * blockDim.x + threadIdx.x;
    if (t >= Bb * N) return;
    int b = t / N, n = t % N;
    const float* p = pts + (size_t)b * 9 * N;
    for (int c = 0; c < 3; ++c) xyz[((size_t)b * 3 + c) * N + n] = p[(size_t)c * N + n];
    for (int c = 0; c < 6; ++c) f[((size_t)b * 6 + c) * N + n]   = p[(size_t)(3 + c) * N + n];
}

// ===================== farthest point sampling =====================
__global__ void k_fps(const float* __restrict__ xyz, float* __restrict__ dist,
                      int* __restrict__ out, int N, int M)
{
    const int b = blockIdx.x, tid = threadIdx.x;
    const float* px = xyz + (size_t)b * 3 * N;
    float* db = dist + (size_t)b * N;
    for (int n = tid; n < N; n += blockDim.x) db[n] = 1e10f;
    __shared__ int sfar;
    __shared__ float sv[256];
    __shared__ int   si[256];
    if (tid == 0) sfar = 0;
    __syncthreads();
    for (int it = 0; it < M; ++it) {
        const int far = sfar;
        if (tid == 0) out[(size_t)b * M + it] = far;
        const float cx = px[far], cy = px[N + far], cz = px[2 * N + far];
        float best = -1.f; int bi = 0x7fffffff;
        for (int n = tid; n < N; n += blockDim.x) {
            float dx = px[n] - cx, dy = px[N + n] - cy, dz = px[2 * N + n] - cz;
            float d  = dx * dx + dy * dy + dz * dz;
            float nd = fminf(db[n], d);
            db[n] = nd;
            if (nd > best) { best = nd; bi = n; }
        }
        sv[tid] = best; si[tid] = bi; __syncthreads();
        for (int st = 128; st > 0; st >>= 1) {
            if (tid < st) {
                if (sv[tid + st] > sv[tid] ||
                    (sv[tid + st] == sv[tid] && si[tid + st] < si[tid])) {
                    sv[tid] = sv[tid + st]; si[tid] = si[tid + st];
                }
            }
            __syncthreads();
        }
        if (tid == 0) sfar = si[0];
        __syncthreads();
    }
}

// new_xyz[b][c][m] = xyz[b][c][fpsidx[b][m]]
__global__ void k_gather_xyz(const float* __restrict__ xyz, const int* __restrict__ idx,
                             float* __restrict__ nxyz, int Bb, int N, int M)
{
    int t = blockIdx.x * blockDim.x + threadIdx.x;
    if (t >= Bb * M) return;
    int b = t / M, m = t % M;
    int id = idx[(size_t)b * M + m];
    for (int c = 0; c < 3; ++c)
        nxyz[((size_t)b * 3 + c) * M + m] = xyz[((size_t)b * 3 + c) * N + id];
}

// first NS indices with d<=r2 (ascending n), padded with the first hit
__global__ void k_query_ball(const float* __restrict__ xyz, const float* __restrict__ nxyz,
                             int* __restrict__ out, int Bb, int N, int M, int NS, float r2)
{
    int t = blockIdx.x * blockDim.x + threadIdx.x;
    if (t >= Bb * M) return;
    int b = t / M, m = t % M;
    const float* px = xyz  + (size_t)b * 3 * N;
    const float* pn = nxyz + (size_t)b * 3 * M;
    float cx = pn[m], cy = pn[M + m], cz = pn[2 * M + m];
    int* o = out + ((size_t)b * M + m) * NS;
    int cnt = 0;
    for (int n = 0; n < N && cnt < NS; ++n) {
        float dx = px[n] - cx, dy = px[N + n] - cy, dz = px[2 * N + n] - cz;
        if (dx * dx + dy * dy + dz * dz <= r2) o[cnt++] = n;
    }
    int first = (cnt > 0) ? o[0] : 0;
    for (; cnt < NS; ++cnt) o[cnt] = first;
}

// h[b][c][k*M+m]: channels 0..2 = grouped coords diff, 3.. = gathered feats
__global__ void k_group(const float* __restrict__ xyz, const float* __restrict__ nxyz,
                        const float* __restrict__ feat, const int* __restrict__ idx,
                        float* __restrict__ h, int Bb, int N, int M, int NS, int Cf)
{
    int t = blockIdx.x * blockDim.x + threadIdx.x;
    if (t >= Bb * NS * M) return;
    int m = t % M, k = (t / M) % NS, b = t / (M * NS);
    int id = idx[((size_t)b * M + m) * NS + k];
    size_t col = (size_t)k * M + m;
    size_t KM  = (size_t)NS * M;
    int Cin = Cf + 3;
    float* hb = h + (size_t)b * Cin * KM;
    for (int c = 0; c < 3; ++c)
        hb[(size_t)c * KM + col] = xyz[((size_t)b * 3 + c) * N + id]
                                 - nxyz[((size_t)b * 3 + c) * M + m];
    for (int c = 0; c < Cf; ++c)
        hb[(size_t)(3 + c) * KM + col] = feat[((size_t)b * Cf + c) * N + id];
}

// out[b][o][m] = max_k in[b][o][k*M+m]
__global__ void k_maxpool(const float* __restrict__ in, float* __restrict__ out,
                          int Bb, int O, int NS, int M)
{
    size_t t = (size_t)blockIdx.x * blockDim.x + threadIdx.x;
    size_t total = (size_t)Bb * O * M;
    if (t >= total) return;
    int m = (int)(t % M); int o = (int)((t / M) % O); int b = (int)(t / ((size_t)O * M));
    const float* p = in + ((size_t)b * O + o) * ((size_t)NS * M);
    float mx = p[m];
    for (int k = 1; k < NS; ++k) mx = fmaxf(mx, p[(size_t)k * M + m]);
    out[((size_t)b * O + o) * M + m] = mx;
}

// ===================== kNN (K=16) for transformer blocks =====================
__global__ void k_knn(const float* __restrict__ xyz, int* __restrict__ out, int Bb, int N)
{
    const int K = 16;
    int t = blockIdx.x * blockDim.x + threadIdx.x;
    if (t >= Bb * N) return;
    int b = t / N, n = t % N;
    const float* px = xyz + (size_t)b * 3 * N;
    float qx = px[n], qy = px[N + n], qz = px[2 * N + n];
    float bd[16]; int bi[16];
    for (int i = 0; i < K; ++i) { bd[i] = 3.0e38f; bi[i] = 0; }
    for (int j = 0; j < N; ++j) {
        float dx = px[j] - qx, dy = px[N + j] - qy, dz = px[2 * N + j] - qz;
        float d = dx * dx + dy * dy + dz * dz;
        if (d < bd[K - 1]) {
            int p = K - 1;
            while (p > 0 && bd[p - 1] > d) { bd[p] = bd[p - 1]; bi[p] = bi[p - 1]; --p; }
            bd[p] = d; bi[p] = j;
        }
    }
    int* o = out + ((size_t)b * N + n) * K;
    for (int i = 0; i < K; ++i) o[i] = bi[i];
}

// logits[b][h][n][kk] = scale * sum_d q[b][h*dh+d][n] * k[b][h*dh+d][idx]
__global__ void k_qk(const float* __restrict__ q, const float* __restrict__ kv,
                     const int* __restrict__ idx, float* __restrict__ logits,
                     int Bb, int C, int N, int H, int K, float scale)
{
    int t = blockIdx.x * blockDim.x + threadIdx.x;
    int total = Bb * H * N * K;
    if (t >= total) return;
    int kk = t % K, n = (t / K) % N, h = (t / (K * N)) % H, b = t / (K * N * H);
    int dh = C / H;
    int id = idx[((size_t)b * N + n) * K + kk];
    const float* qb = q  + (size_t)b * C * N;
    const float* kb = kv + (size_t)b * 2 * C * N;     // k = rows [0,C)
    float s = 0.f;
    for (int d = 0; d < dh; ++d) {
        int c = h * dh + d;
        s += qb[(size_t)c * N + n] * kb[(size_t)c * N + id];
    }
    logits[t] = s * scale;
}

// rel[b][c][kk*N+n] = xyz[b][c][idx[b][n][kk]] - xyz[b][c][n]
__global__ void k_rel(const float* __restrict__ xyz, const int* __restrict__ idx,
                      float* __restrict__ rel, int Bb, int N, int K)
{
    int t = blockIdx.x * blockDim.x + threadIdx.x;
    if (t >= Bb * K * N) return;
    int n = t % N, kk = (t / N) % K, b = t / (N * K);
    int id = idx[((size_t)b * N + n) * K + kk];
    size_t KN = (size_t)K * N;
    for (int c = 0; c < 3; ++c)
        rel[(size_t)b * 3 * KN + (size_t)c * KN + (size_t)kk * N + n] =
            xyz[((size_t)b * 3 + c) * N + id] - xyz[((size_t)b * 3 + c) * N + n];
}

// rb[b][h][n][kk] = sum_d r2[b][h*dh+d][kk*N+n]
__global__ void k_rb(const float* __restrict__ r2, float* __restrict__ rb,
                     int Bb, int C, int N, int H, int K)
{
    int t = blockIdx.x * blockDim.x + threadIdx.x;
    int total = Bb * H * N * K;
    if (t >= total) return;
    int kk = t % K, n = (t / K) % N, h = (t / (K * N)) % H, b = t / (K * N * H);
    int dh = C / H;
    size_t KN = (size_t)K * N;
    const float* p = r2 + (size_t)b * C * KN + (size_t)h * dh * KN + (size_t)kk * N + n;
    float s = 0.f;
    for (int d = 0; d < dh; ++d) s += p[(size_t)d * KN];
    rb[t] = s;
}

// attn = softmax over K of (logits + rb); in-place into logits
__global__ void k_softmax(float* __restrict__ logits, const float* __restrict__ rb,
                          int Bb, int H, int N, int K)
{
    int t = blockIdx.x * blockDim.x + threadIdx.x;
    if (t >= Bb * H * N) return;
    float* p = logits + (size_t)t * K;
    const float* r = rb + (size_t)t * K;
    float v[16];
    float mx = -3.0e38f;
    for (int i = 0; i < K; ++i) { v[i] = p[i] + r[i]; mx = fmaxf(mx, v[i]); }
    float s = 0.f;
    for (int i = 0; i < K; ++i) { v[i] = __expf(v[i] - mx); s += v[i]; }
    float inv = 1.0f / s;
    for (int i = 0; i < K; ++i) p[i] = v[i] * inv;
}

// out[b][c][n] = sum_kk v[b][C+c][idx] * attn[b][h(c)][n][kk]
__global__ void k_attn_out(const float* __restrict__ kv, const float* __restrict__ attn,
                           const int* __restrict__ idx, float* __restrict__ out,
                           int Bb, int C, int N, int H, int K)
{
    int t = blockIdx.x * blockDim.x + threadIdx.x;
    int total = Bb * C * N;
    if (t >= total) return;
    int n = t % N, c = (t / N) % C, b = t / (C * N);
    int h = c / (C / H);
    const float* vb = kv + (size_t)b * 2 * C * N + (size_t)(C + c) * N;
    const float* ab = attn + (((size_t)b * H + h) * N + n) * K;
    const int*   ib = idx + ((size_t)b * N + n) * K;
    float s = 0.f;
    for (int kk = 0; kk < K; ++kk) s += vb[ib[kk]] * ab[kk];
    out[((size_t)b * C + c) * N + n] = s;
}

// dst[b][coff+c][n] = src[b][c][n]   (dst has Cin channels)
__global__ void k_copych(const float* __restrict__ src, float* __restrict__ dst,
                         int Bb, int Csrc, int N, int Cin, int coff)
{
    size_t t = (size_t)blockIdx.x * blockDim.x + threadIdx.x;
    size_t total = (size_t)Bb * Csrc * N;
    if (t >= total) return;
    int n = (int)(t % N); int c = (int)((t / N) % Csrc); int b = (int)(t / ((size_t)Csrc * N));
    dst[((size_t)b * Cin + coff + c) * N + n] = src[t];
}

// 3-NN interpolation: h[b][coff+c][n1] = sum_j w_j * p2[b][c][id_j]
__global__ void k_three(const float* __restrict__ xyz1, const float* __restrict__ xyz2,
                        const float* __restrict__ p2, float* __restrict__ h,
                        int Bb, int N1, int N2, int C2, int coff, int Cin)
{
    int t = blockIdx.x * blockDim.x + threadIdx.x;
    if (t >= Bb * N1) return;
    int b = t / N1, n = t % N1;
    const float* px1 = xyz1 + (size_t)b * 3 * N1;
    const float* px2 = xyz2 + (size_t)b * 3 * N2;
    float qx = px1[n], qy = px1[N1 + n], qz = px1[2 * N1 + n];
    float d0 = 3.0e38f, d1 = 3.0e38f, d2 = 3.0e38f;
    int i0 = 0, i1 = 0, i2 = 0;
    for (int j = 0; j < N2; ++j) {
        float dx = px2[j] - qx, dy = px2[N2 + j] - qy, dz = px2[2 * N2 + j] - qz;
        float d = dx * dx + dy * dy + dz * dz;
        if (d < d0)      { d2 = d1; i2 = i1; d1 = d0; i1 = i0; d0 = d; i0 = j; }
        else if (d < d1) { d2 = d1; i2 = i1; d1 = d; i1 = j; }
        else if (d < d2) { d2 = d; i2 = j; }
    }
    float w0 = 1.0f / (d0 + 1e-8f), w1 = 1.0f / (d1 + 1e-8f), w2 = 1.0f / (d2 + 1e-8f);
    float inv = 1.0f / (w0 + w1 + w2);
    w0 *= inv; w1 *= inv; w2 *= inv;
    const float* pb = p2 + (size_t)b * C2 * N2;
    float* hb = h + ((size_t)b * Cin + coff) * N1;
    for (int c = 0; c < C2; ++c)
        hb[(size_t)c * N1 + n] = w0 * pb[(size_t)c * N2 + i0]
                               + w1 * pb[(size_t)c * N2 + i1]
                               + w2 * pb[(size_t)c * N2 + i2];
}

// out[b][n][O] = x[b][o][n]
__global__ void k_transpose_out(const float* __restrict__ x, float* __restrict__ out,
                                int Bb, int O, int N)
{
    size_t t = (size_t)blockIdx.x * blockDim.x + threadIdx.x;
    size_t total = (size_t)Bb * O * N;
    if (t >= total) return;
    int n = (int)(t % N); int o = (int)((t / N) % O); int b = (int)(t / ((size_t)O * N));
    out[((size_t)b * N + n) * O + o] = x[t];
}

// =====================================================================
// Host orchestration
// =====================================================================
extern "C" void kernel_launch(void* const* d_in, const int* in_sizes, int n_in,
                              void* d_out, int out_size, void* d_ws, size_t ws_size,
                              hipStream_t stream)
{
    (void)in_sizes; (void)n_in; (void)out_size; (void)ws_size;
    const int BS = 8, N0 = 8192, HH = 4, NS = 32;

    // ---- parameter walker (setup_inputs dict insertion order) ----
    int pi = 1;  // d_in[0] = points
    auto NP = [&]() { return (const float*)d_in[pi++]; };

    struct MLPL { const float *W, *g, *b; int o, c; };
    auto readMLP = [&](MLPL* L, int cin, const int* outs, int nl) {
        for (int i = 0; i < nl; ++i) {
            L[i].W = NP(); L[i].g = NP(); L[i].b = NP();
            L[i].o = outs[i]; L[i].c = cin; cin = outs[i];
        }
    };
    MLPL sa1L[3], sa2L[3], sa3L[3], sa4L[3];
    { int o[3] = {32, 32, 64};    readMLP(sa1L, 9,   o, 3); }
    { int o[3] = {64, 64, 128};   readMLP(sa2L, 67,  o, 3); }
    { int o[3] = {128, 128, 256}; readMLP(sa3L, 131, o, 3); }
    { int o[3] = {256, 256, 512}; readMLP(sa4L, 259, o, 3); }

    struct PEP { const float *w1, *g, *b, *w2, *b2; int hid, o; };
    auto readPE = [&](int hid, int o) {
        PEP p; p.w1 = NP(); p.g = NP(); p.b = NP(); p.w2 = NP(); p.b2 = NP();
        p.hid = hid; p.o = o; return p;
    };
    PEP pe2 = readPE(64, 128);
    PEP pe3 = readPE(128, 256);

    struct BLK { const float *q, *kv, *proj, *rp_w1, *rp_g, *rp_b, *rp_w2,
                             *n1_g, *n1_b, *n2_g, *n2_b, *ffn_w1, *ffn_b1, *ffn_w2, *ffn_b2; };
    auto readBlk = [&]() {
        BLK k;
        k.q = NP(); k.kv = NP(); k.proj = NP(); k.rp_w1 = NP(); k.rp_g = NP(); k.rp_b = NP();
        k.rp_w2 = NP(); k.n1_g = NP(); k.n1_b = NP(); k.n2_g = NP(); k.n2_b = NP();
        k.ffn_w1 = NP(); k.ffn_b1 = NP(); k.ffn_w2 = NP(); k.ffn_b2 = NP();
        return k;
    };
    BLK tr2b[2]; tr2b[0] = readBlk(); tr2b[1] = readBlk();
    BLK tr3b[2]; tr3b[0] = readBlk(); tr3b[1] = readBlk();

    MLPL fp4L[2], fp3L[2], fp2L[2], fp1L[3];
    { int o[2] = {256, 256}; readMLP(fp4L, 768, o, 2); }
    { int o[2] = {256, 256}; readMLP(fp3L, 384, o, 2); }
    { int o[2] = {256, 128}; readMLP(fp2L, 320, o, 2); }
    { int o[3] = {128, 128, 128}; readMLP(fp1L, 134, o, 3); }
    const float* conv1_w = NP();
    const float* bn1_g   = NP();
    const float* bn1_b   = NP();
    const float* conv2_w = NP();
    const float* conv2_b = NP();

    const float* points = (const float*)d_in[0];

    // ---- workspace bump allocator ----
    char* wsb = (char*)d_ws;
    size_t off = 0;
    auto AF = [&](size_t n) { float* r = (float*)(wsb + off);
                              off += ((n * sizeof(float)) + 255) & ~(size_t)255; return r; };
    auto AI = [&](size_t n) { int* r = (int*)(wsb + off);
                              off += ((n * sizeof(int)) + 255) & ~(size_t)255; return r; };

    // persistent tensors
    float* xyz0 = AF((size_t)BS * 3 * N0);
    float* f0   = AF((size_t)BS * 6 * N0);
    float* xyz1 = AF((size_t)BS * 3 * 1024); float* l1 = AF((size_t)BS * 64 * 1024);
    float* xyz2 = AF((size_t)BS * 3 * 256);  float* l2 = AF((size_t)BS * 128 * 256);
    float* xyz3 = AF((size_t)BS * 3 * 64);   float* l3 = AF((size_t)BS * 256 * 64);
    float* xyz4 = AF((size_t)BS * 3 * 16);   float* l4 = AF((size_t)BS * 512 * 16);
    float* l3n  = AF((size_t)BS * 256 * 64);
    float* l2n  = AF((size_t)BS * 256 * 256);
    float* l1n  = AF((size_t)BS * 128 * 1024);
    float* gmean = AF(1024);
    float* gvar  = AF(1024);
    float* fpsdist = AF((size_t)BS * N0);
    int*   idxbuf  = AI((size_t)BS * 1024 * 32);  // ball-query / knn scratch
    int*   fpsidx  = AI((size_t)BS * 1024);

    auto GEMM = [&](const float* W, const float* X, const float* bias, float* Y,
                    int O, int C, int Ncol, int relu) {
        if (((O & 31) == 0) && ((Ncol & 31) == 0)) {
            long waves = (long)BS * (O >> 5) * (Ncol >> 5);
            long blocks = (waves * 32 + 127) / 128;
            k_gemm_wmma_f<<<dim3((unsigned)blocks), dim3(128), 0, stream>>>(
                W, X, bias, Y, BS, O, C, Ncol, relu);
        } else {
            long waves = (long)BS * ((O + 15) >> 4) * ((Ncol + 31) >> 5);
            long blocks = (waves * 32 + 127) / 128;
            k_gemm_wmma<<<dim3((unsigned)blocks), dim3(128), 0, stream>>>(
                W, X, bias, Y, BS, O, C, Ncol, relu);
        }
    };
    auto BN = [&](float* X, const float* g, const float* b, int O, int Ncol, int relu) {
        k_bn_stats<<<dim3(O), dim3(256), 0, stream>>>(X, gmean, gvar, BS, O, Ncol);
        size_t tot = (size_t)BS * O * Ncol;
        k_bn_apply<<<dim3((unsigned)((tot + 255) / 256)), dim3(256), 0, stream>>>(
            X, gmean, gvar, g, b, BS, O, Ncol, relu);
    };
    auto ADD = [&](float* y, const float* x, size_t n) {
        k_add<<<dim3((unsigned)((n + 255) / 256)), dim3(256), 0, stream>>>(y, x, n);
    };

    // ---- set abstraction ----
    auto SA = [&](MLPL* L, int nl, int Npt, int M, float radius, int Cf,
                  const float* xin, const float* fin, float* xout, float* fout) {
        size_t save = off;
        k_fps<<<dim3(BS), dim3(256), 0, stream>>>(xin, fpsdist, fpsidx, Npt, M);
        k_gather_xyz<<<dim3((BS * M + 255) / 256), dim3(256), 0, stream>>>(xin, fpsidx, xout, BS, Npt, M);
        k_query_ball<<<dim3((BS * M + 255) / 256), dim3(256), 0, stream>>>(
            xin, xout, idxbuf, BS, Npt, M, NS, radius * radius);
        int KM = NS * M;
        float* h = AF((size_t)BS * (Cf + 3) * KM);
        k_group<<<dim3((BS * NS * M + 255) / 256), dim3(256), 0, stream>>>(
            xin, xout, fin, idxbuf, h, BS, Npt, M, NS, Cf);
        float* cur = h; int cc = Cf + 3;
        for (int i = 0; i < nl; ++i) {
            float* t = AF((size_t)BS * L[i].o * KM);
            GEMM(L[i].W, cur, nullptr, t, L[i].o, cc, KM, 0);
            BN(t, L[i].g, L[i].b, L[i].o, KM, 1);
            cur = t; cc = L[i].o;
        }
        size_t tot = (size_t)BS * cc * M;
        k_maxpool<<<dim3((unsigned)((tot + 255) / 256)), dim3(256), 0, stream>>>(cur, fout, BS, cc, NS, M);
        off = save;
    };

    auto POSMLP = [&](PEP& p, const float* xyz, float* feat, int M) {
        size_t save = off;
        float* t1 = AF((size_t)BS * p.hid * M);
        GEMM(p.w1, xyz, nullptr, t1, p.hid, 3, M, 0);
        BN(t1, p.g, p.b, p.hid, M, 1);
        float* t2 = AF((size_t)BS * p.o * M);
        GEMM(p.w2, t1, p.b2, t2, p.o, p.hid, M, 0);
        ADD(feat, t2, (size_t)BS * p.o * M);
        off = save;
    };

    auto TRBLOCK = [&](BLK& bk, float* x, const float* xyz, int C, int Npt, const int* knn) {
        size_t save = off;
        const int K = 16;
        float scale = 1.0f / sqrtf((float)(C / HH));
        float* q  = AF((size_t)BS * C * Npt);
        GEMM(bk.q, x, nullptr, q, C, C, Npt, 0);
        float* kv = AF((size_t)BS * 2 * C * Npt);
        GEMM(bk.kv, x, nullptr, kv, 2 * C, C, Npt, 0);
        float* logits = AF((size_t)BS * HH * Npt * K);
        k_qk<<<dim3((BS * HH * Npt * K + 255) / 256), dim3(256), 0, stream>>>(
            q, kv, knn, logits, BS, C, Npt, HH, K, scale);
        int KN = K * Npt;
        float* rel = AF((size_t)BS * 3 * KN);
        k_rel<<<dim3((BS * KN + 255) / 256), dim3(256), 0, stream>>>(xyz, knn, rel, BS, Npt, K);
        float* r1 = AF((size_t)BS * C * KN);
        GEMM(bk.rp_w1, rel, nullptr, r1, C, 3, KN, 0);
        BN(r1, bk.rp_g, bk.rp_b, C, KN, 1);
        float* r2 = AF((size_t)BS * C * KN);
        GEMM(bk.rp_w2, r1, nullptr, r2, C, C, KN, 0);
        float* rb = AF((size_t)BS * HH * Npt * K);
        k_rb<<<dim3((BS * HH * Npt * K + 255) / 256), dim3(256), 0, stream>>>(r2, rb, BS, C, Npt, HH, K);
        k_softmax<<<dim3((BS * HH * Npt + 255) / 256), dim3(256), 0, stream>>>(logits, rb, BS, HH, Npt, K);
        float* ao = AF((size_t)BS * C * Npt);
        k_attn_out<<<dim3((BS * C * Npt + 255) / 256), dim3(256), 0, stream>>>(
            kv, logits, knn, ao, BS, C, Npt, HH, K);
        float* po = AF((size_t)BS * C * Npt);
        GEMM(bk.proj, ao, nullptr, po, C, C, Npt, 0);
        ADD(x, po, (size_t)BS * C * Npt);
        BN(x, bk.n1_g, bk.n1_b, C, Npt, 0);
        float* ff1 = AF((size_t)BS * 4 * C * Npt);
        GEMM(bk.ffn_w1, x, bk.ffn_b1, ff1, 4 * C, C, Npt, 1);
        float* ff2 = AF((size_t)BS * C * Npt);
        GEMM(bk.ffn_w2, ff1, bk.ffn_b2, ff2, C, 4 * C, Npt, 0);
        ADD(x, ff2, (size_t)BS * C * Npt);
        BN(x, bk.n2_g, bk.n2_b, C, Npt, 0);
        off = save;
    };

    auto FP = [&](MLPL* L, int nl, const float* x1, const float* x2,
                  const float* p1, int C1, const float* p2, int C2,
                  int N1, int N2, float* outbuf) {
        size_t save = off;
        int Cin = C1 + C2;
        float* h = AF((size_t)BS * Cin * N1);
        size_t ctot = (size_t)BS * C1 * N1;
        k_copych<<<dim3((unsigned)((ctot + 255) / 256)), dim3(256), 0, stream>>>(p1, h, BS, C1, N1, Cin, 0);
        k_three<<<dim3((BS * N1 + 255) / 256), dim3(256), 0, stream>>>(x1, x2, p2, h, BS, N1, N2, C2, C1, Cin);
        float* cur = h; int cc = Cin;
        for (int i = 0; i < nl; ++i) {
            float* t = (i == nl - 1) ? outbuf : AF((size_t)BS * L[i].o * N1);
            GEMM(L[i].W, cur, nullptr, t, L[i].o, cc, N1, 0);
            BN(t, L[i].g, L[i].b, L[i].o, N1, 1);
            cur = t; cc = L[i].o;
        }
        off = save;
    };

    // ============== forward pass ==============
    k_split<<<dim3((BS * N0 + 255) / 256), dim3(256), 0, stream>>>(points, xyz0, f0, BS, N0);

    SA(sa1L, 3, N0,   1024, 0.1f, 6,   xyz0, f0, xyz1, l1);
    SA(sa2L, 3, 1024, 256,  0.2f, 64,  xyz1, l1, xyz2, l2);
    SA(sa3L, 3, 256,  64,   0.4f, 128, xyz2, l2, xyz3, l3);
    SA(sa4L, 3, 64,   16,   0.8f, 256, xyz3, l3, xyz4, l4);

    POSMLP(pe2, xyz2, l2, 256);
    k_knn<<<dim3((BS * 256 + 255) / 256), dim3(256), 0, stream>>>(xyz2, idxbuf, BS, 256);
    TRBLOCK(tr2b[0], l2, xyz2, 128, 256, idxbuf);
    TRBLOCK(tr2b[1], l2, xyz2, 128, 256, idxbuf);

    POSMLP(pe3, xyz3, l3, 64);
    k_knn<<<dim3((BS * 64 + 255) / 256), dim3(256), 0, stream>>>(xyz3, idxbuf, BS, 64);
    TRBLOCK(tr3b[0], l3, xyz3, 256, 64, idxbuf);
    TRBLOCK(tr3b[1], l3, xyz3, 256, 64, idxbuf);

    FP(fp4L, 2, xyz3, xyz4, l3, 256, l4,  512, 64,   16,   l3n);
    FP(fp3L, 2, xyz2, xyz3, l2, 128, l3n, 256, 256,  64,   l2n);
    FP(fp2L, 2, xyz1, xyz2, l1, 64,  l2n, 256, 1024, 256,  l1n);

    {   // fp1 + segmentation head
        size_t save = off;
        float* l0 = AF((size_t)BS * 128 * N0);
        FP(fp1L, 3, xyz0, xyz1, f0, 6, l1n, 128, N0, 1024, l0);
        float* t = AF((size_t)BS * 128 * N0);
        GEMM(conv1_w, l0, nullptr, t, 128, 128, N0, 0);
        BN(t, bn1_g, bn1_b, 128, N0, 1);
        float* o13 = AF((size_t)BS * 13 * N0);
        GEMM(conv2_w, t, conv2_b, o13, 13, 128, N0, 0);
        size_t tot = (size_t)BS * 13 * N0;
        k_transpose_out<<<dim3((unsigned)((tot + 255) / 256)), dim3(256), 0, stream>>>(
            o13, (float*)d_out, BS, 13, N0);
        off = save;
    }
}
